// GDSW_71519795413401
// MI455X (gfx1250) — matrix operand order repaired and checked by reading, not compiled, not run on hardware
//
#include <hip/hip_runtime.h>
#include <hip/hip_bf16.h>
#include <math.h>

typedef __bf16 bf16_t;
typedef __attribute__((ext_vector_type(16))) __bf16 v16bf;
typedef __attribute__((ext_vector_type(8)))  __bf16 v8bf;
typedef __attribute__((ext_vector_type(8)))  float  v8f;

// Problem constants
#define B_    256
#define T_    64
#define BT    (B_*T_)          // 16384
#define KPAD  768              // 763 padded to 768 (multiple of 32)
#define NOUT  640              // 128 (a1) + 128 (o1) + 384 (ih)
#define HID_  128
#define GEMB_ 720

// ---------------- workspace layout (bytes) ----------------
#define WS_U     0
#define WS_WCAT  25165824
#define WS_WHH   26148864
#define WS_DEMO  26247168
#define WS_C     26257408

// ---------------- CDNA5 async global->LDS helpers ----------------
#if defined(__has_builtin)
#if __has_builtin(__builtin_amdgcn_global_load_async_to_lds_b128)
#define HAS_ASYNC_LDS 1
#endif
#if __has_builtin(__builtin_amdgcn_s_wait_asynccnt)
#define HAS_WAIT_ASYNC 1
#endif
#endif
#ifndef HAS_ASYNC_LDS
#define HAS_ASYNC_LDS 0
#endif
#ifndef HAS_WAIT_ASYNC
#define HAS_WAIT_ASYNC 0
#endif

// Builtin expects: (int4 AS(1)*, int4 AS(3)*, imm offset, imm cpol)
typedef int v4i_vs __attribute__((vector_size(16)));
typedef __attribute__((address_space(1))) v4i_vs as1_v4i;
typedef __attribute__((address_space(3))) v4i_vs as3_v4i;

__device__ __forceinline__ void async_copy16(const void* g, void* l) {
#if HAS_ASYNC_LDS
    __builtin_amdgcn_global_load_async_to_lds_b128((as1_v4i*)g, (as3_v4i*)l, 0, 0);
#else
    *(v8bf*)l = *(const v8bf*)g;   // sync fallback through VGPRs
#endif
}

template <int N>
__device__ __forceinline__ void wait_async() {
#if HAS_WAIT_ASYNC
    __builtin_amdgcn_s_wait_asynccnt((short)N);
#elif HAS_ASYNC_LDS
    asm volatile("s_wait_asynccnt %0" :: "n"(N) : "memory");
#endif
}

__device__ __forceinline__ float elu_f(float s) {
    return s > 0.f ? s : (expf(s) - 1.f);
}

// ------------------------------------------------------------------
// Pack masked/reordered weights to bf16 (Wcat matches U column layout).
// ------------------------------------------------------------------
__global__ void k_prep(const float* __restrict__ Wa1, const float* __restrict__ Wo1,
                       const float* __restrict__ Wih, const float* __restrict__ Whh,
                       bf16_t* __restrict__ Wcat, bf16_t* __restrict__ WhhB) {
    int i = blockIdx.x * blockDim.x + threadIdx.x;
    if (i < NOUT * KPAD) {
        int row = i / KPAD, col = i % KPAD;
        float w = 0.f;
        if (row < 128) {
            if (col >= 32 && col < 42)        w = Wa1[row * 730 + (col - 32)];
            else if (col >= 43 && col < 763)  w = Wa1[row * 730 + 10 + (col - 43)];
        } else if (row < 256) {
            int r = row - 128;
            if (col < 763)                    w = Wo1[r * 763 + col];
        } else {
            int r = row - 256;
            if (col < 32)                     w = Wih[r * 753 + col];
            else if (col == 42)               w = Wih[r * 753 + 32];
            else if (col >= 43 && col < 763)  w = Wih[r * 753 + 33 + (col - 43)];
        }
        Wcat[i] = (bf16_t)w;
    } else {
        int j = i - NOUT * KPAD;
        if (j < 384 * 128) WhhB[j] = (bf16_t)Whh[j];
    }
}

// demo = x_demo @ W_stat.T + b_stat   (256 x 10)
__global__ void k_demo(const float* __restrict__ xd, const float* __restrict__ Ws,
                       const float* __restrict__ bs, float* __restrict__ demo) {
    int i = blockIdx.x * blockDim.x + threadIdx.x;
    if (i >= B_ * 10) return;
    int b = i / 10, j = i % 10;
    float s = bs[j];
    for (int k = 0; k < 10; k++) s += xd[b * 10 + k] * Ws[j * 10 + k];
    demo[i] = s;
}

// ------------------------------------------------------------------
// Per (b,t): node MLP -> edge MLP -> gemb; xemb; assemble bf16 row of U:
//   U layout: [0:32 xemb][32:42 demo][42 treat][43:763 gemb][763:768 zero]
// ------------------------------------------------------------------
__global__ __launch_bounds__(128) void k_feat(
        const float* __restrict__ x, const float* __restrict__ ftr,
        const float* __restrict__ demo,
        const float* __restrict__ Wx2e, const float* __restrict__ bx2e,
        const float* __restrict__ Wm1a, const float* __restrict__ bm1a,
        const float* __restrict__ Wm1b, const float* __restrict__ bm1b,
        const float* __restrict__ Wm2a, const float* __restrict__ bm2a,
        const float* __restrict__ Wm2b, const float* __restrict__ bm2b,
        bf16_t* __restrict__ U) {
    const int bt = blockIdx.x;          // row = b*T + t
    const int b  = bt >> 6;
    const int t  = bt & 63;
    const int tid = threadIdx.x;

    __shared__ float xs[44];
    __shared__ float h1a[80];
    __shared__ float h1b[80];
    __shared__ float h2a[GEMB_];

    const float* xr = x + ((size_t)b * (T_ + 1) + t) * 44;
    if (tid < 44) xs[tid] = xr[tid];
    __syncthreads();

    if (tid < 80) {                      // node MLP layer 1: 4 -> 8, elu
        int ag = tid >> 3, j = tid & 7;
        float s = bm1a[j];
        for (int f = 0; f < 4; f++) s += xs[ag * 4 + f] * Wm1a[j * 4 + f];
        h1a[tid] = elu_f(s);
    }
    __syncthreads();
    if (tid < 80) {                      // node MLP layer 2: 8 -> 8, elu
        int ag = tid >> 3, j = tid & 7;
        float s = bm1b[j];
        for (int k = 0; k < 8; k++) s += h1a[ag * 8 + k] * Wm1b[j * 8 + k];
        h1b[tid] = elu_f(s);
    }
    __syncthreads();
    for (int task = tid; task < GEMB_; task += 128) {   // edge MLP layer 1
        int e = task >> 3, j = task & 7;
        int recv = e / 9;
        int k9   = e - recv * 9;
        int send = k9 + (k9 >= recv ? 1 : 0);
        float s = bm2a[j];
        for (int k = 0; k < 8; k++) s += h1b[recv * 8 + k] * Wm2a[j * 16 + k];
        for (int k = 0; k < 8; k++) s += h1b[send * 8 + k] * Wm2a[j * 16 + 8 + k];
        h2a[task] = elu_f(s);
    }
    __syncthreads();

    bf16_t* Ur = U + (size_t)bt * KPAD;
    for (int task = tid; task < GEMB_; task += 128) {   // edge MLP layer 2 -> gemb
        int e = task >> 3, j = task & 7;
        float s = bm2b[j];
        for (int k = 0; k < 8; k++) s += h2a[e * 8 + k] * Wm2b[j * 8 + k];
        Ur[43 + task] = (bf16_t)elu_f(s);
    }
    if (tid < 32) {                      // xemb
        float s = bx2e[tid];
        for (int f = 0; f < 4; f++) s += xs[40 + f] * Wx2e[tid * 4 + f];
        Ur[tid] = (bf16_t)s;
    }
    if (tid >= 32 && tid < 42) Ur[tid] = (bf16_t)demo[b * 10 + (tid - 32)];
    if (tid == 42)             Ur[42]  = (bf16_t)ftr[(size_t)b * (T_ + 1) + t];
    if (tid >= 43 && tid < 48) Ur[720 + tid] = (bf16_t)0.f;   // pad 763..767
}

// ------------------------------------------------------------------
// Fragment loaders following ISA §7.12.2 layouts (wave32).
// A (16-bit 16x32, MxK): lane<16 -> K{0..7,16..23}; lane>=16 -> K{8..15,24..31}
// B (16-bit 32x16, KxN): lane<16 -> K{0..7,16..23}; lane>=16 -> K{8..15,24..31} of column n
// ------------------------------------------------------------------
__device__ __forceinline__ v16bf load_a_frag(const bf16_t* row_base, int koff /*0|8*/) {
    v8bf lo = *(const v8bf*)(row_base + koff);
    v8bf hi = *(const v8bf*)(row_base + koff + 16);
    return __builtin_shufflevector(lo, hi, 0,1,2,3,4,5,6,7,8,9,10,11,12,13,14,15);
}
__device__ __forceinline__ v16bf load_b_frag(const bf16_t* row_base, int kb /*0|8*/) {
    v8bf lo = *(const v8bf*)(row_base + kb);
    v8bf hi = *(const v8bf*)(row_base + kb + 16);
    return __builtin_shufflevector(lo, hi, 0,1,2,3,4,5,6,7,8,9,10,11,12,13,14,15);
}

// ------------------------------------------------------------------
// C[16384][640] = U[16384][768] @ Wcat[640][768]^T via bf16 WMMA.
// Double-buffered LDS tiles filled with async global->LDS copies.
// Block: 128(M) x 128(N), 8 waves as 4(M) x 2(N); wave tile 32x64 (2x4 frags).
// ------------------------------------------------------------------
__global__ __launch_bounds__(256) void k_gemm(const bf16_t* __restrict__ U,
                                              const bf16_t* __restrict__ W,
                                              float* __restrict__ C) {
    __shared__ bf16_t sA[2][128][32];   // 2 x 8KB
    __shared__ bf16_t sB[2][128][32];   // 2 x 8KB

    const int mBase = blockIdx.x * 128;
    const int nBase = blockIdx.y * 128;
    const int tid   = threadIdx.x;
    const int lane  = tid & 31;
    const int wave  = tid >> 5;
    const int wm    = wave & 3;        // 0..3 (M)
    const int wn    = wave >> 2;       // 0..1 (N)

    v8f acc[2][4];
    for (int i = 0; i < 2; i++)
        for (int j = 0; j < 4; j++) acc[i][j] = {};

    const int lm   = lane & 15;
    const int koff = (lane >> 4) << 3;   // A frag
    const int kb   = (lane >> 4) << 3;   // B frag

    // per-thread copy slots: 2 x v128 for A, 2 x v128 for B per k-step
    const int r0 = tid >> 2;             // row for slot 0 (vid=tid)
    const int c0 = (tid & 3) << 3;
    const int r1 = (tid + 256) >> 2;     // row for slot 1
    const int c1 = ((tid + 256) & 3) << 3;

#define ISSUE_TILE(buf, k0)                                                        \
    do {                                                                           \
        async_copy16(&U[(size_t)(mBase + r0) * KPAD + (k0) + c0], &sA[buf][r0][c0]); \
        async_copy16(&U[(size_t)(mBase + r1) * KPAD + (k0) + c1], &sA[buf][r1][c1]); \
        async_copy16(&W[(size_t)(nBase + r0) * KPAD + (k0) + c0], &sB[buf][r0][c0]); \
        async_copy16(&W[(size_t)(nBase + r1) * KPAD + (k0) + c1], &sB[buf][r1][c1]); \
    } while (0)

    ISSUE_TILE(0, 0);

    const int KT = KPAD / 32;            // 24
    for (int kt = 0; kt < KT; kt++) {
        const int cur = kt & 1;
        if (kt + 1 < KT) {
            ISSUE_TILE(cur ^ 1, (kt + 1) * 32);   // overlap next tile with compute
            wait_async<4>();                      // current tile's 4 ops complete
        } else {
            wait_async<0>();
        }
        __syncthreads();

        v16bf aF[2], bF[4];
        for (int i = 0; i < 2; i++)
            aF[i] = load_a_frag(&sA[cur][wm * 32 + i * 16 + lm][0], koff);
        for (int j = 0; j < 4; j++)
            bF[j] = load_b_frag(&sB[cur][wn * 64 + j * 16 + lm][0], kb);

        for (int i = 0; i < 2; i++)
            for (int j = 0; j < 4; j++)
                acc[i][j] = __builtin_amdgcn_wmma_f32_16x16x32_bf16(
                    false, aF[i], false, bF[j], (short)0, acc[i][j], false, false);
        __syncthreads();                 // buffer 'cur' free for reuse
    }
#undef ISSUE_TILE

    // D layout: m = (lane>>4)*8 + r, n = lane&15
    const int m0 = (lane >> 4) << 3;
    for (int i = 0; i < 2; i++)
        for (int j = 0; j < 4; j++) {
            int n  = nBase + wn * 64 + j * 16 + lm;
            int mr = mBase + wm * 32 + i * 16 + m0;
            for (int r = 0; r < 8; r++)
                C[(size_t)(mr + r) * NOUT + n] = acc[i][j][r];
        }
}

// ------------------------------------------------------------------
// a_out / y_out heads: relu(C + bias) dot W2, one block per (b,t) row.
// ------------------------------------------------------------------
__global__ __launch_bounds__(128) void k_epi(const float* __restrict__ C,
                                             const float* __restrict__ ba1, const float* __restrict__ Wa2,
                                             const float* __restrict__ bo1, const float* __restrict__ Wo2,
                                             float* __restrict__ out) {
    const int row = blockIdx.x;
    const int t   = threadIdx.x;
    __shared__ float red[128];

    float ca = C[(size_t)row * NOUT + t] + ba1[t];
    float va = (ca > 0.f ? ca : 0.f) * Wa2[t];
    float co = C[(size_t)row * NOUT + 128 + t] + bo1[t];
    float vo = (co > 0.f ? co : 0.f) * Wo2[t];

    red[t] = va; __syncthreads();
    for (int s = 64; s > 0; s >>= 1) { if (t < s) red[t] += red[t + s]; __syncthreads(); }
    if (t == 0) out[(size_t)row * 130 + 0] = red[0];
    __syncthreads();
    red[t] = vo; __syncthreads();
    for (int s = 64; s > 0; s >>= 1) { if (t < s) red[t] += red[t + s]; __syncthreads(); }
    if (t == 0) out[(size_t)row * 130 + 1] = red[0];
}

// ------------------------------------------------------------------
// GRU scan. 16 blocks x 16 batch rows; h in LDS (fp32 + bf16 copy);
// W_hh (bf16) preloaded to LDS once via async copies (96KB of 320KB/WGP).
// Per step: gh = h_bf(16x128) @ Whh^T via WMMA, then fp32 gate math.
// gi rows for step t+1 are prefetched (global_prefetch) during step t.
// ------------------------------------------------------------------
__global__ __launch_bounds__(256) void k_gru(const float* __restrict__ C,
                                             const bf16_t* __restrict__ WhhB,
                                             const float* __restrict__ h0,
                                             const float* __restrict__ bih,
                                             const float* __restrict__ bhh,
                                             float* __restrict__ out) {
    const int g   = blockIdx.x;        // 0..15
    const int b0  = g * 16;
    const int tid = threadIdx.x;
    const int lane = tid & 31;
    const int wave = tid >> 5;

    __shared__ bf16_t wS[384 * HID_];   // 96KB resident W_hh
    __shared__ float  hS[16 * HID_];    // fp32 state
    __shared__ bf16_t hB[16 * HID_];    // bf16 copy for WMMA
    __shared__ float  gh[16 * 384];

    for (int i = tid * 8; i < 384 * HID_; i += 256 * 8)
        async_copy16(&WhhB[i], &wS[i]);

    for (int i = tid; i < 16 * HID_; i += 256) {
        int rr = i >> 7, j = i & 127;
        float v = h0[(size_t)(b0 + rr) * HID_ + j];
        hS[i] = v; hB[i] = (bf16_t)v;
    }
    wait_async<0>();
    __syncthreads();

    const int lm   = lane & 15;
    const int koff = (lane >> 4) << 3;
    const int kb   = (lane >> 4) << 3;
    const int m0   = (lane >> 4) << 3;

    for (int t = 0; t < T_; t++) {
        // gh = hB @ Whh^T : M=16 (1 frag), N=384 (24 frags, 3 per wave), K=128
        for (int nf = wave * 3; nf < wave * 3 + 3; nf++) {
            v8f acc = {};
            for (int kk = 0; kk < 4; kk++) {
                v16bf aF = load_a_frag(&hB[lm * HID_ + kk * 32], koff);
                v16bf bF = load_b_frag(&wS[(nf * 16 + lm) * HID_ + kk * 32], kb);
                acc = __builtin_amdgcn_wmma_f32_16x16x32_bf16(
                    false, aF, false, bF, (short)0, acc, false, false);
            }
            for (int r = 0; r < 8; r++)
                gh[(m0 + r) * 384 + nf * 16 + lm] = acc[r];
        }
        __syncthreads();

        // gate math (fp32)
        for (int i = tid; i < 16 * HID_; i += 256) {
            int rr = i >> 7, j = i & 127;
            int b  = b0 + rr;
            size_t crow = ((size_t)b * T_ + t) * NOUT;
            if (t + 1 < T_ && (j & 15) == 0) {   // prefetch next step's gi rows
                const float* nxt = &C[((size_t)b * T_ + (t + 1)) * NOUT + 256 + j];
                __builtin_prefetch(nxt, 0, 3);
                __builtin_prefetch(nxt + 128, 0, 3);
                __builtin_prefetch(nxt + 256, 0, 3);
            }
            float ir  = C[crow + 256 + j]       + bih[j];
            float iz  = C[crow + 384 + j]       + bih[128 + j];
            float in_ = C[crow + 512 + j]       + bih[256 + j];
            float hr  = gh[rr * 384 + j]        + bhh[j];
            float hz  = gh[rr * 384 + 128 + j]  + bhh[128 + j];
            float hn  = gh[rr * 384 + 256 + j]  + bhh[256 + j];
            float r = 1.f / (1.f + expf(-(ir + hr)));
            float z = 1.f / (1.f + expf(-(iz + hz)));
            float n = tanhf(in_ + r * hn);
            float hnew = (1.f - z) * n + z * hS[i];
            hS[i] = hnew;
            hB[i] = (bf16_t)hnew;
            out[((size_t)b * T_ + t) * 130 + 2 + j] = hnew;
        }
        __syncthreads();
    }
}

extern "C" void kernel_launch(void* const* d_in, const int* in_sizes, int n_in,
                              void* d_out, int out_size, void* d_ws, size_t ws_size,
                              hipStream_t stream) {
    (void)in_sizes; (void)n_in; (void)out_size; (void)ws_size;
    const float* x      = (const float*)d_in[0];
    const float* x_demo = (const float*)d_in[1];
    const float* ftr    = (const float*)d_in[2];
    const float* h0     = (const float*)d_in[3];
    const float* Wx2e   = (const float*)d_in[4];
    const float* bx2e   = (const float*)d_in[5];
    const float* Wstat  = (const float*)d_in[6];
    const float* bstat  = (const float*)d_in[7];
    const float* Wm1a   = (const float*)d_in[8];
    const float* bm1a   = (const float*)d_in[9];
    const float* Wm1b   = (const float*)d_in[10];
    const float* bm1b   = (const float*)d_in[11];
    const float* Wm2a   = (const float*)d_in[12];
    const float* bm2a   = (const float*)d_in[13];
    const float* Wm2b   = (const float*)d_in[14];
    const float* bm2b   = (const float*)d_in[15];
    const float* Wa1    = (const float*)d_in[16];
    const float* ba1    = (const float*)d_in[17];
    const float* Wa2    = (const float*)d_in[18];
    const float* Wo1    = (const float*)d_in[19];
    const float* bo1    = (const float*)d_in[20];
    const float* Wo2    = (const float*)d_in[21];
    const float* Wih    = (const float*)d_in[22];
    const float* bih    = (const float*)d_in[23];
    const float* Whh    = (const float*)d_in[24];
    const float* bhh    = (const float*)d_in[25];
    float* out = (float*)d_out;

    char* ws = (char*)d_ws;
    bf16_t* U    = (bf16_t*)(ws + WS_U);
    bf16_t* Wcat = (bf16_t*)(ws + WS_WCAT);
    bf16_t* WhhB = (bf16_t*)(ws + WS_WHH);
    float*  demo = (float*) (ws + WS_DEMO);
    float*  C    = (float*) (ws + WS_C);

    int prepN = NOUT * KPAD + 384 * 128;
    k_prep<<<(prepN + 255) / 256, 256, 0, stream>>>(Wa1, Wo1, Wih, Whh, Wcat, WhhB);
    k_demo<<<(B_ * 10 + 255) / 256, 256, 0, stream>>>(x_demo, Wstat, bstat, demo);
    k_feat<<<BT, 128, 0, stream>>>(x, ftr, demo, Wx2e, bx2e,
                                   Wm1a, bm1a, Wm1b, bm1b, Wm2a, bm2a, Wm2b, bm2b, U);
    k_gemm<<<dim3(BT / 128, NOUT / 128), 256, 0, stream>>>(U, Wcat, C);
    k_epi<<<BT, 128, 0, stream>>>(C, ba1, Wa2, bo1, Wo2, out);
    k_gru<<<16, 256, 0, stream>>>(C, WhhB, h0, bih, bhh, out);
}
